// ContrastiveLoss2_35124242546892
// MI455X (gfx1250) — matrix operand order repaired and checked by reading, not compiled, not run on hardware
//
#include <hip/hip_runtime.h>

// NT-Xent contrastive loss, fused Gram + row-softmax-denominator on gfx1250 WMMA,
// with double-buffered async global->LDS staging of the B panel.
// B=4096, D=256, N=2B=8192, tau=0.5, eps=1e-8.

#define BROWS 4096
#define NROWS 8192
#define DDIM 256
#define ROWS_PER_WG 64
#define COLSPLIT 8
#define COLS_PER_SPLIT (NROWS / COLSPLIT) // 1024
#define NPARTS (COLSPLIT * 2)             // 16 partial buffers (split x ct)
#define TAU_INV 2.0f
#define EPSN 1e-8f

typedef __attribute__((ext_vector_type(16))) _Float16 v16h;
typedef __attribute__((ext_vector_type(8)))  _Float16 v8h;
typedef __attribute__((ext_vector_type(8)))  float    v8f;

union Frag { v16h v; v8h h[2]; };

// ---------------- 1) row L2-normalize: z (f32) -> zn (f16) + 1/norm (f32) --------
__global__ __launch_bounds__(256) void nt_xent_normalize(
    const float* __restrict__ z_orig, const float* __restrict__ z_aug,
    _Float16* __restrict__ znh, float* __restrict__ invnorm) {
  const int row = blockIdx.x;   // 0..NROWS-1 ; z = cat([z_augment, z_orig])
  const int t = threadIdx.x;    // 0..255 == DDIM
  const float* src = (row < BROWS) ? (z_aug + (size_t)row * DDIM)
                                   : (z_orig + (size_t)(row - BROWS) * DDIM);
  float x = src[t];
  __shared__ float red[256];
  red[t] = x * x;
  __syncthreads();
  for (int s = 128; s > 0; s >>= 1) {
    if (t < s) red[t] += red[t + s];
    __syncthreads();
  }
  float inv = 1.0f / fmaxf(sqrtf(red[0]), EPSN);
  if (t == 0) invnorm[row] = inv;
  znh[(size_t)row * DDIM + t] = (_Float16)(x * inv);
}

// ---------------- 2) positives in full f32: pos_i = <zn_i, zn_{p(i)}> ------------
__global__ __launch_bounds__(256) void nt_xent_pos(
    const float* __restrict__ z_orig, const float* __restrict__ z_aug,
    const float* __restrict__ invnorm, float* __restrict__ pos) {
  const int i = blockIdx.x;
  const int t = threadIdx.x;
  const int p = (i < BROWS) ? (i + BROWS) : (i - BROWS);
  const float* zi = (i < BROWS) ? (z_aug + (size_t)i * DDIM)
                                : (z_orig + (size_t)(i - BROWS) * DDIM);
  const float* zp = (p < BROWS) ? (z_aug + (size_t)p * DDIM)
                                : (z_orig + (size_t)(p - BROWS) * DDIM);
  __shared__ float red[256];
  red[t] = zi[t] * zp[t];
  __syncthreads();
  for (int s = 128; s > 0; s >>= 1) {
    if (t < s) red[t] += red[t + s];
    __syncthreads();
  }
  if (t == 0) pos[i] = red[0] * invnorm[i] * invnorm[p];
}

// ---------------- 3) fused Gram + sum_{j!=i} exp(S_ij/tau) via WMMA --------------
// Block = 256 thr = 8 waves: wave = (row-tile rt 0..3, col-parity ct 0..1).
// Grid = (NROWS/64 row blocks, COLSPLIT column splits).
// B panel (32 columns = 32 contiguous zn rows = 16 KB) is staged into LDS with
// async global->LDS copies, double buffered one iteration ahead (ASYNCcnt).
__global__ __launch_bounds__(256) void nt_xent_gram(
    const _Float16* __restrict__ znh, float* __restrict__ part) {
  __shared__ __attribute__((aligned(16))) _Float16 sB[2][32 * DDIM]; // 2 x 16 KB

  const int tid = threadIdx.x;
  const int wave = tid >> 5;
  const int lane = tid & 31;
  const int rt = wave >> 1;
  const int ct = wave & 1;
  const int halfSel = lane >> 4;   // which half-wave
  const int l16 = lane & 15;
  const int rowbase = blockIdx.x * ROWS_PER_WG;
  const int split = blockIdx.y;
  const int cbase = split * COLS_PER_SPLIT;

  // --- preload A fragments (16 rows x 256 K) once; reused for all column tiles ---
  // A layout (ISA 7.12.2, 16-bit A 16x32): lane m=l16; lanes0-15 hold K 0..7 &
  // 16..23, lanes16-31 hold K 8..15 & 24..31 (two contiguous 16B chunks).
  const int arow = rowbase + rt * 16 + l16;
  const _Float16* aptr = znh + (size_t)arow * DDIM;
  Frag a[DDIM / 32];
#pragma unroll
  for (int kf = 0; kf < DDIM / 32; ++kf) {
    a[kf].h[0] = *(const v8h*)(aptr + kf * 32 + halfSel * 8);
    a[kf].h[1] = *(const v8h*)(aptr + kf * 32 + 16 + halfSel * 8);
  }

  // per-thread async-copy addresses: thread t moves 64 bytes of the 16 KB strip.
  // Generic LDS pointers carry the raw LDS byte offset in bits [31:0] (ISA 10.2),
  // which is exactly what VDST of global_load_async_to_lds expects.
  unsigned lbase[2];
  lbase[0] = (unsigned)(uintptr_t)(&sB[0][0]) + (unsigned)tid * 64u;
  lbase[1] = (unsigned)(uintptr_t)(&sB[1][0]) + (unsigned)tid * 64u;

  auto issue = [&](int it, int buf) {
    unsigned long long ga =
        (unsigned long long)(uintptr_t)(znh + (size_t)(cbase + it * 32) * DDIM) +
        (unsigned long long)((unsigned)tid * 64u);
    unsigned la = lbase[buf];
    // INST_OFFSET is applied to both the LDS and the global address (ISA 8/4.4).
    asm volatile(
        "global_load_async_to_lds_b128 %0, %1, off\n\t"
        "global_load_async_to_lds_b128 %0, %1, off offset:16\n\t"
        "global_load_async_to_lds_b128 %0, %1, off offset:32\n\t"
        "global_load_async_to_lds_b128 %0, %1, off offset:48"
        :: "v"(la), "v"(ga) : "memory");
  };

  float racc[8];
#pragma unroll
  for (int v = 0; v < 8; ++v) racc[v] = 0.0f;

  const int iters = COLS_PER_SPLIT / 32;    // 32
  issue(0, 0);

  for (int it = 0; it < iters; ++it) {
    const int buf = it & 1;
    if (it + 1 < iters) {
      issue(it + 1, buf ^ 1);               // prefetch next panel
      asm volatile("s_wait_asynccnt 0x4" ::: "memory"); // this panel's 4 done
    } else {
      asm volatile("s_wait_asynccnt 0x0" ::: "memory");
    }
    __syncthreads();                        // whole cooperative panel visible

    // B layout (32x16): lane holds column n=l16; lanes0-15 K=0..7 & 16..23,
    // lanes16-31 K=8..15 & 24..31 -> two 16B chunks of the zn row (now in LDS).
    const _Float16* bls = &sB[buf][(ct * 16 + l16) * DDIM];
    v8f c = {};
#pragma unroll
    for (int kf = 0; kf < DDIM / 32; ++kf) {
      Frag b;
      b.h[0] = *(const v8h*)(bls + kf * 32 + halfSel * 8);
      b.h[1] = *(const v8h*)(bls + kf * 32 + 16 + halfSel * 8);
      c = __builtin_amdgcn_wmma_f32_16x16x32_f16(
          /*neg_a=*/false, a[kf].v, /*neg_b=*/false, b.v,
          /*c_mod=*/(short)0, c, /*reuse_a=*/false, /*reuse_b=*/false);
    }
    // epilogue: C/D layout: lane l: col n=l16, row m = halfSel*8 + v
    const int gc = cbase + it * 32 + ct * 16 + l16;
#pragma unroll
    for (int v = 0; v < 8; ++v) {
      const int gr = rowbase + rt * 16 + halfSel * 8 + v;
      float e = (gr == gc) ? 0.0f : __expf(c[v] * TAU_INV);
      racc[v] += e;  // per-lane register accumulation: deterministic order
    }
    __syncthreads();  // everyone done reading sB[buf] before it is refilled
  }

  // one cross-lane reduction at the end (16 lanes share each row)
#pragma unroll
  for (int v = 0; v < 8; ++v) {
    float r = racc[v];
    r += __shfl_xor(r, 1, 16);
    r += __shfl_xor(r, 2, 16);
    r += __shfl_xor(r, 4, 16);
    r += __shfl_xor(r, 8, 16);
    racc[v] = r;
  }
  if (l16 == 0) {  // lanes 0 and 16 own rows m and m+8
    const int pslot = split * 2 + ct;  // private partial buffer -> no atomics
#pragma unroll
    for (int v = 0; v < 8; ++v) {
      const int gr = rowbase + rt * 16 + halfSel * 8 + v;
      part[(size_t)pslot * NROWS + gr] = racc[v];
    }
  }
}

// ---------------- 4) loss = mean_i [ log(sum_parts) - pos_i/tau ] ----------------
__global__ __launch_bounds__(256) void nt_xent_loss(
    const float* __restrict__ part, const float* __restrict__ pos,
    float* __restrict__ out) {
  const int t = threadIdx.x;
  float acc = 0.0f;
  for (int i = t; i < NROWS; i += 256) {
    float s = 0.0f;
#pragma unroll
    for (int p = 0; p < NPARTS; ++p) s += part[(size_t)p * NROWS + i];
    acc += logf(s) - pos[i] * TAU_INV;
  }
  __shared__ float red[256];
  red[t] = acc;
  __syncthreads();
  for (int s = 128; s > 0; s >>= 1) {
    if (t < s) red[t] += red[t + s];
    __syncthreads();
  }
  if (t == 0) out[0] = red[0] / (float)NROWS;
}

extern "C" void kernel_launch(void* const* d_in, const int* in_sizes, int n_in,
                              void* d_out, int out_size, void* d_ws, size_t ws_size,
                              hipStream_t stream) {
  (void)in_sizes; (void)n_in; (void)out_size; (void)ws_size;
  const float* z_orig = (const float*)d_in[0];
  const float* z_aug  = (const float*)d_in[1];
  float* out = (float*)d_out;

  char* ws = (char*)d_ws;
  _Float16* znh = (_Float16*)ws;
  size_t off = (size_t)NROWS * DDIM * sizeof(_Float16);   // 4 MB
  float* invnorm = (float*)(ws + off); off += (size_t)NROWS * sizeof(float);
  float* pos     = (float*)(ws + off); off += (size_t)NROWS * sizeof(float);
  float* part    = (float*)(ws + off);                    // NPARTS*NROWS floats

  nt_xent_normalize<<<NROWS, 256, 0, stream>>>(z_orig, z_aug, znh, invnorm);
  nt_xent_pos<<<NROWS, 256, 0, stream>>>(z_orig, z_aug, invnorm, pos);
  dim3 grid(NROWS / ROWS_PER_WG, COLSPLIT);
  nt_xent_gram<<<grid, 256, 0, stream>>>(znh, part);
  nt_xent_loss<<<1, 256, 0, stream>>>(part, pos, out);
}
